// MultiHead_4827543241010
// MI455X (gfx1250) — matrix operand order repaired and checked by reading, not compiled, not run on hardware
//
#include <hip/hip_runtime.h>
#include <hip/hip_bf16.h>

// ---------------------------------------------------------------------------
// MultiHead attention forward, bf16 WMMA + TDM pipeline for gfx1250 (MI455X).
// B=256, T=256, C=384, H=6, D=64.  scale = C^-0.5 (reference quirk).
// ---------------------------------------------------------------------------

#define BB 256
#define TT 256
#define CC 384
#define HH 6
#define DD 64

typedef __bf16 bf16;
typedef __attribute__((ext_vector_type(16))) __bf16   bf16x16;
typedef __attribute__((ext_vector_type(8)))  float    f32x8;
typedef __attribute__((ext_vector_type(4)))  unsigned u32x4;
typedef __attribute__((ext_vector_type(8)))  unsigned u32x8;

union Frag16 {
    uint4   u[2];
    bf16x16 v;
};

// Load a 16-element bf16 fragment from two 16-byte chunks (generic ptr: works
// for both global and LDS address spaces after inference).
__device__ __forceinline__ bf16x16 load_frag(const bf16* p0, const bf16* p1) {
    Frag16 f;
    f.u[0] = *(const uint4*)p0;
    f.u[1] = *(const uint4*)p1;
    return f.v;
}

__device__ __forceinline__ f32x8 wmma_bf16(bf16x16 a, bf16x16 b, f32x8 c) {
    // D = A(16x32 bf16) * B(32x16 bf16) + C(16x16 f32)
    return __builtin_amdgcn_wmma_f32_16x16x32_bf16(
        false, a, false, b, (short)0, c, false, false);
}

// ---------------------------------------------------------------------------
// Tensor Data Mover: load a 2D bf16 tile (tile1 rows x tile0 elems, row stride
// stride0 elems) from global memory into LDS at byte offset lds_off.
// D# layout per CDNA5 ISA ch.8: group0 = {count, lds_addr, global_addr, type=2},
// group1 = {data_size=2B, tensor dims, tile dims, dim0 stride}.
// Issued by one wave; completion via TENSORcnt.
// ---------------------------------------------------------------------------
__device__ __forceinline__ void tdm_load_2d_bf16(
    unsigned lds_off, const bf16* gptr,
    unsigned dim0, unsigned dim1, unsigned tile0, unsigned tile1,
    unsigned stride0) {
    const unsigned long long ga = (unsigned long long)(uintptr_t)gptr;
    u32x4 g0;
    g0[0] = 1u;                                   // count=1, user descriptor
    g0[1] = lds_off;                              // lds_addr (bytes)
    g0[2] = (unsigned)ga;                         // global_addr[31:0]
    g0[3] = ((unsigned)(ga >> 32) & 0x01FFFFFFu)  // global_addr[56:32]
            | (2u << 30);                         // type=2 (image)
    u32x8 g1;
    g1[0] = 0x00010000u;                          // data_size=1 (2 bytes)
    g1[1] = (dim0 & 0xFFFFu) << 16;               // tensor_dim0[15:0]
    g1[2] = ((dim0 >> 16) & 0xFFFFu) | ((dim1 & 0xFFFFu) << 16);
    g1[3] = ((dim1 >> 16) & 0xFFFFu) | (tile0 << 16);  // tile_dim0
    g1[4] = tile1;                                // tile_dim1 (tile_dim2=0)
    g1[5] = stride0;                              // tensor_dim0_stride[31:0]
    g1[6] = 0u;
    g1[7] = 0u;
    asm volatile("tensor_load_to_lds %0, %1" :: "s"(g0), "s"(g1) : "memory");
}

// ---------------------------------------------------------------------------
// Conversion kernels (fp32 -> bf16, with layout change for weights)
// ---------------------------------------------------------------------------

__global__ void cvt_plain(const float* __restrict__ s, bf16* __restrict__ d, long n) {
    long i = (long)blockIdx.x * blockDim.x + threadIdx.x;
    long stride = (long)gridDim.x * blockDim.x;
    for (; i < n; i += stride) d[i] = (bf16)s[i];
}

// Wq/Wk/Wv: [H,C,D] fp32 -> WT[h*D+d][c] bf16  (B^T layout for the GEMM)
__global__ void cvt_wqkv(const float* __restrict__ s, bf16* __restrict__ d) {
    int i = blockIdx.x * blockDim.x + threadIdx.x;
    if (i >= HH * CC * DD) return;
    int h = i / (CC * DD);
    int r = i % (CC * DD);
    int c = r / DD;
    int dd = r % DD;
    d[(size_t)(h * DD + dd) * CC + c] = (bf16)s[i];
}

// Wp: [C,C] fp32 -> WpT[c2][c1] bf16
__global__ void cvt_wp(const float* __restrict__ s, bf16* __restrict__ d) {
    int i = blockIdx.x * blockDim.x + threadIdx.x;
    if (i >= CC * CC) return;
    int c1 = i / CC, c2 = i % CC;
    d[(size_t)c2 * CC + c1] = (bf16)s[i];
}

// ---------------------------------------------------------------------------
// GEMM:  D[M,N] = A[M,K] * BT[N,K]^T   (K = N = 384, M = B*T = 65536)
// Block tile 128x64, 8 waves, wave tile 32x32 (2x2 WMMA tiles), K-loop 12x32.
// The 64x384 B^T panel is staged into LDS once per workgroup by the TDM and
// consumed by all 8 waves (4x global-traffic reduction on the panel).
// MODE 0: store bf16 row-major [M,N]          (q, k)
// MODE 1: store bf16 transposed as [B,H,D,T]  (v)
// MODE 2: store fp32 + bias to output         (final projection)
// ---------------------------------------------------------------------------

template <int MODE>
__global__ __launch_bounds__(256) void gemm_kernel(
    const bf16* __restrict__ A, const bf16* __restrict__ BT,
    void* __restrict__ Dst, const float* __restrict__ bias) {
    const int lane = threadIdx.x & 31;
    const int w    = threadIdx.x >> 5;
    const int ln   = lane & 15;
    const int hi   = lane >> 4;   // 0 or 1 (lane half)
    const int hi8  = hi * 8;
    const int Mbase = blockIdx.x * 128 + (w & 3) * 32;
    const int Nloc  = (w >> 2) * 32;             // N offset inside LDS panel
    const int Nbase = blockIdx.y * 64 + Nloc;
    const int K = CC, N = CC;

    // B^T panel for this workgroup: 64 rows (output cols) x 384 (K), 48 KB.
    __shared__ __align__(16) bf16 ldsB[64 * CC];

    if (w == 0) {
        tdm_load_2d_bf16((unsigned)(uintptr_t)&ldsB[0],
                         BT + (size_t)(blockIdx.y * 64) * K,
                         /*dim0=*/K, /*dim1=*/64, /*tile0=*/K, /*tile1=*/64,
                         /*stride0=*/K);
        __builtin_amdgcn_s_wait_tensorcnt(0);
    }
    __syncthreads();

    f32x8 acc[2][2] = {};

#pragma unroll 1
    for (int kb = 0; kb < 12; ++kb) {
        const int k0 = kb * 32;
        bf16x16 a[2], b[2];
#pragma unroll
        for (int mt = 0; mt < 2; ++mt) {
            // A fragment (16x32): row = lane%16, K = {hi8..hi8+7, hi8+16..hi8+23}
            const bf16* p = A + (size_t)(Mbase + mt * 16 + ln) * K + k0 + hi8;
            a[mt] = load_frag(p, p + 16);
        }
#pragma unroll
        for (int nt = 0; nt < 2; ++nt) {
            // B fragment (32x16): col = lane%16, K = {hi8..hi8+7, hi8+16..hi8+23}
            const bf16* p = ldsB + (size_t)(Nloc + nt * 16 + ln) * K + k0 + hi8;
            b[nt] = load_frag(p, p + 16);
        }
#pragma unroll
        for (int mt = 0; mt < 2; ++mt)
#pragma unroll
            for (int nt = 0; nt < 2; ++nt)
                acc[mt][nt] = wmma_bf16(a[mt], b[nt], acc[mt][nt]);
    }

#pragma unroll
    for (int mt = 0; mt < 2; ++mt)
#pragma unroll
        for (int nt = 0; nt < 2; ++nt)
#pragma unroll
            for (int v = 0; v < 8; ++v) {
                const int row = Mbase + mt * 16 + v + hi8;
                const int col = Nbase + nt * 16 + ln;
                const float val = acc[mt][nt][v];
                if (MODE == 0) {
                    ((bf16*)Dst)[(size_t)row * N + col] = (bf16)val;
                } else if (MODE == 1) {
                    const int bb = row >> 8, t = row & 255;
                    const int h = col >> 6, d = col & 63;
                    ((bf16*)Dst)[(((size_t)bb * HH + h) * DD + d) * TT + t] = (bf16)val;
                } else {
                    ((float*)Dst)[(size_t)row * N + col] = val + bias[col];
                }
            }
}

// ---------------------------------------------------------------------------
// Fused causal attention, flash-style online softmax.
// One workgroup per (b,h); wave w owns q rows [32w, 32w+32); causal -> wave w
// iterates key blocks j = 0..w (32 keys each).
// The strided K head-slice [T x 64] (row stride 384) is gathered into LDS by
// one TDM op per workgroup; all waves consume it via aligned ds_load_b128.
// qf,kf: [B*T, C] bf16 (head slice at col h*64);  vt: [B,H,D,T] bf16.
// ---------------------------------------------------------------------------

__global__ __launch_bounds__(256) void attn_kernel(
    const bf16* __restrict__ qf, const bf16* __restrict__ kf,
    const bf16* __restrict__ vt, bf16* __restrict__ attn) {
    const int bh = blockIdx.x;
    const int b = bh / HH, h = bh % HH;
    const int lane = threadIdx.x & 31;
    const int w    = threadIdx.x >> 5;
    const int ln   = lane & 15;
    const int hi   = lane >> 4;
    const int hi8  = hi * 8;
    const int qb   = w * 32;

    // K panel for this (b,h): 256 keys x 64 dims, 32 KB (TDM-gathered).
    __shared__ __align__(16) bf16 ldsK[TT * DD];
    // Per-wave P staging tile: 32 x 32, row stride 40 bf16 (80 B: keeps
    // ds_load_b128 16-byte aligned and skews banks).  20 KB.
    __shared__ __align__(16) bf16 ldsP[8 * 32 * 40];
    bf16* plds = &ldsP[w * 32 * 40];

    if (w == 0) {
        tdm_load_2d_bf16((unsigned)(uintptr_t)&ldsK[0],
                         kf + (size_t)(b * TT) * CC + h * DD,
                         /*dim0=*/DD, /*dim1=*/TT, /*tile0=*/DD, /*tile1=*/TT,
                         /*stride0=*/CC);
        __builtin_amdgcn_s_wait_tensorcnt(0);
    }
    __syncthreads();

    // Q fragments: 2 M-tiles x 2 K-chunks, loaded once (each row read once).
    bf16x16 qa[2][2];
#pragma unroll
    for (int mt = 0; mt < 2; ++mt)
#pragma unroll
        for (int kc = 0; kc < 2; ++kc) {
            const bf16* p = qf + (size_t)(b * TT + qb + mt * 16 + ln) * CC
                               + h * DD + kc * 32 + hi8;
            qa[mt][kc] = load_frag(p, p + 16);
        }

    f32x8 o[2][4] = {};
    float m_s[2][8], l_s[2][8];
#pragma unroll
    for (int mt = 0; mt < 2; ++mt)
#pragma unroll
        for (int v = 0; v < 8; ++v) { m_s[mt][v] = -1e30f; l_s[mt][v] = 0.0f; }

    const float scale = 0.051031036307982884f;  // 384^-0.5 (reference quirk)

    for (int j = 0; j <= w; ++j) {
        // ---- S = Q @ K^T for this 32x32 tile block (K from LDS) ----
        f32x8 s[2][2] = {};
#pragma unroll
        for (int kc = 0; kc < 2; ++kc) {
            bf16x16 bk[2];
#pragma unroll
            for (int nt = 0; nt < 2; ++nt) {
                const int srow = j * 32 + nt * 16 + ln;  // key index
                const bf16* p = ldsK + srow * DD + kc * 32 + hi8;
                bk[nt] = load_frag(p, p + 16);
            }
#pragma unroll
            for (int mt = 0; mt < 2; ++mt)
#pragma unroll
                for (int nt = 0; nt < 2; ++nt)
                    s[mt][nt] = wmma_bf16(qa[mt][kc], bk[nt], s[mt][nt]);
        }

        // ---- scale, causal mask, online softmax, P -> LDS ----
#pragma unroll
        for (int mt = 0; mt < 2; ++mt) {
#pragma unroll
            for (int v = 0; v < 8; ++v) {
                const int r = qb + mt * 16 + v + hi8;
                float x0 = s[mt][0][v] * scale;
                float x1 = s[mt][1][v] * scale;
                if (j * 32 + ln > r)      x0 = -1e30f;
                if (j * 32 + 16 + ln > r) x1 = -1e30f;
                // row max over 16 lanes of this half + both n-tiles
                float t = fmaxf(x0, x1);
#pragma unroll
                for (int off = 1; off < 16; off <<= 1)
                    t = fmaxf(t, __shfl_xor(t, off, 32));
                const float mn = fmaxf(m_s[mt][v], t);
                const float alpha = __expf(m_s[mt][v] - mn);
                m_s[mt][v] = mn;
                const float p0 = __expf(x0 - mn);
                const float p1 = __expf(x1 - mn);
                float rs = p0 + p1;
#pragma unroll
                for (int off = 1; off < 16; off <<= 1)
                    rs += __shfl_xor(rs, off, 32);
                l_s[mt][v] = l_s[mt][v] * alpha + rs;
#pragma unroll
                for (int nt = 0; nt < 4; ++nt) o[mt][nt][v] *= alpha;
                const int row = mt * 16 + v + hi8;
                plds[row * 40 + ln]      = (bf16)p0;
                plds[row * 40 + 16 + ln] = (bf16)p1;
            }
        }

        // ---- O += P @ V (reload P as A-fragments from LDS) ----
        bf16x16 pa[2];
#pragma unroll
        for (int mt = 0; mt < 2; ++mt) {
            const bf16* p = plds + (mt * 16 + ln) * 40 + hi8;
            pa[mt] = load_frag(p, p + 16);  // wave-private LDS: DScnt-ordered
        }
#pragma unroll
        for (int nt = 0; nt < 4; ++nt) {
            const int d = nt * 16 + ln;
            const bf16* p = vt + ((size_t)(b * HH + h) * DD + d) * TT
                               + j * 32 + hi8;
            bf16x16 bv = load_frag(p, p + 16);  // two 16 B chunks global
#pragma unroll
            for (int mt = 0; mt < 2; ++mt)
                o[mt][nt] = wmma_bf16(pa[mt], bv, o[mt][nt]);
        }
    }

    // ---- normalize and store concat-head output [B*T, C] bf16 ----
#pragma unroll
    for (int mt = 0; mt < 2; ++mt)
#pragma unroll
        for (int v = 0; v < 8; ++v) {
            const float inv = 1.0f / l_s[mt][v];
            const int r = qb + mt * 16 + v + hi8;
#pragma unroll
            for (int nt = 0; nt < 4; ++nt) {
                const int col = h * DD + nt * 16 + ln;
                attn[(size_t)(b * TT + r) * CC + col] = (bf16)(o[mt][nt][v] * inv);
            }
        }
}

// ---------------------------------------------------------------------------
// Host launcher
// ---------------------------------------------------------------------------

extern "C" void kernel_launch(void* const* d_in, const int* in_sizes, int n_in,
                              void* d_out, int out_size, void* d_ws, size_t ws_size,
                              hipStream_t stream) {
    const float* x  = (const float*)d_in[0];
    const float* Wq = (const float*)d_in[1];
    const float* Wk = (const float*)d_in[2];
    const float* Wv = (const float*)d_in[3];
    const float* Wp = (const float*)d_in[4];
    const float* bp = (const float*)d_in[5];
    float* out = (float*)d_out;

    const size_t XB = (size_t)BB * TT * CC;  // 25,165,824 elements
    char* ws = (char*)d_ws;
    size_t off = 0;
    auto carve = [&](size_t bytes) -> char* {
        char* p = ws + off;
        off += (bytes + 255) & ~(size_t)255;
        return p;
    };
    bf16* xb    = (bf16*)carve(XB * 2);
    bf16* wqT   = (bf16*)carve((size_t)CC * CC * 2);
    bf16* wkT   = (bf16*)carve((size_t)CC * CC * 2);
    bf16* wvT   = (bf16*)carve((size_t)CC * CC * 2);
    bf16* wpT   = (bf16*)carve((size_t)CC * CC * 2);
    bf16* qf    = (bf16*)carve(XB * 2);
    bf16* kf    = (bf16*)carve(XB * 2);
    bf16* vt    = (bf16*)carve(XB * 2);
    bf16* attnb = (bf16*)carve(XB * 2);
    (void)ws_size; (void)in_sizes; (void)n_in; (void)out_size;

    // 1) precision conversion + weight layout transforms
    cvt_plain<<<2048, 256, 0, stream>>>(x, xb, (long)XB);
    cvt_wqkv<<<(HH * CC * DD + 255) / 256, 256, 0, stream>>>(Wq, wqT);
    cvt_wqkv<<<(HH * CC * DD + 255) / 256, 256, 0, stream>>>(Wk, wkT);
    cvt_wqkv<<<(HH * CC * DD + 255) / 256, 256, 0, stream>>>(Wv, wvT);
    cvt_wp  <<<(CC * CC + 255) / 256, 256, 0, stream>>>(Wp, wpT);

    // 2) QKV projections: [65536,384] @ [384,384]
    dim3 ggrid(BB * TT / 128, CC / 64);
    gemm_kernel<0><<<ggrid, 256, 0, stream>>>(xb, wqT, qf, nullptr);
    gemm_kernel<0><<<ggrid, 256, 0, stream>>>(xb, wkT, kf, nullptr);
    gemm_kernel<1><<<ggrid, 256, 0, stream>>>(xb, wvT, vt, nullptr);

    // 3) fused causal attention, one WG per (b,h)
    attn_kernel<<<BB * HH, 256, 0, stream>>>(qf, kf, vt, attnb);

    // 4) output projection with bias, fp32 result
    gemm_kernel<2><<<ggrid, 256, 0, stream>>>(attnb, wpT, out, bp);
}
